// CGCLayer_33724083208218
// MI455X (gfx1250) — compile-verified, matrix operand order checked by reading, not run on hardware
//
#include <hip/hip_runtime.h>
#include <hip/hip_bf16.h>

typedef __attribute__((ext_vector_type(16))) _Float16 v16h;
typedef __attribute__((ext_vector_type(8)))  float    v8f;

constexpr int kB    = 4096;
constexpr int kDin  = 1024;
constexpr int kH1   = 1024;
constexpr int kH2   = 512;
constexpr int kE    = 8;   // 6 specific + 2 shared experts
constexpr int kSpec = 6;

union Frag { uint4 u[2]; v16h h; };

// CDNA5 async global->LDS path (guarded; falls back to sync copies)
#if defined(__AMDGCN__) && \
    __has_builtin(__builtin_amdgcn_global_load_async_to_lds_b128) && \
    __has_builtin(__builtin_amdgcn_s_wait_asynccnt)
#define USE_ASYNC 1
typedef int v4i __attribute__((ext_vector_type(4)));
typedef __attribute__((address_space(3))) v4i lds_v4i;
typedef __attribute__((address_space(1))) v4i glb_v4i;
#else
#define USE_ASYNC 0
#endif

// ---------------- f32 -> f16 conversion (x4 vectorized) ----------------
__global__ void __launch_bounds__(256)
cvt_f32_to_f16(const float* __restrict__ s, _Float16* __restrict__ d, int n4) {
  int i = blockIdx.x * blockDim.x + threadIdx.x;
  if (i >= n4) return;
  float4 v = reinterpret_cast<const float4*>(s)[i];
  union { _Float16 h[4]; uint2 u; } p;
  p.h[0] = (_Float16)v.x; p.h[1] = (_Float16)v.y;
  p.h[2] = (_Float16)v.z; p.h[3] = (_Float16)v.w;
  reinterpret_cast<uint2*>(d)[i] = p.u;
}

// ------------- f32 [E][K][N] -> f16 [E][N][K] transpose+convert --------
__global__ void __launch_bounds__(256)
cvt_transpose_f32_f16(const float* __restrict__ in, _Float16* __restrict__ out,
                      int K, int N) {
  __shared__ float tile[32][33];
  const int e  = blockIdx.z;
  const int n0 = blockIdx.x * 32;
  const int k0 = blockIdx.y * 32;
  const int tx = threadIdx.x;   // 32
  const int ty = threadIdx.y;   // 8
  const float* src = in + (size_t)e * K * N;
  _Float16* dst = out + (size_t)e * N * K;
  #pragma unroll
  for (int i = 0; i < 4; i++)
    tile[ty + i * 8][tx] = src[(size_t)(k0 + ty + i * 8) * N + (n0 + tx)];
  __syncthreads();
  #pragma unroll
  for (int i = 0; i < 4; i++)
    dst[(size_t)(n0 + ty + i * 8) * K + (k0 + tx)] = (_Float16)tile[tx][ty + i * 8];
}

// ---------------- gate weights: [B,20] = softmax groups (4,4,4,8) ------
__global__ void __launch_bounds__(128)
gates_kernel(const float* __restrict__ x0, const float* __restrict__ x1,
             const float* __restrict__ x2, const float* __restrict__ xs,
             const float* __restrict__ Wg,  const float* __restrict__ bg,
             const float* __restrict__ Wsg, const float* __restrict__ bsg,
             float* __restrict__ gates) {
  __shared__ float xr[4][kDin];
  __shared__ float lg[20];
  const int row = blockIdx.x;
  const int t = threadIdx.x;
  for (int i = t; i < kDin; i += 128) {
    xr[0][i] = x0[(size_t)row * kDin + i];
    xr[1][i] = x1[(size_t)row * kDin + i];
    xr[2][i] = x2[(size_t)row * kDin + i];
    xr[3][i] = xs[(size_t)row * kDin + i];
  }
  __syncthreads();
  if (t < 20) {
    float s;
    if (t < 12) {                       // per-domain gates: Linear(D_IN, 4)
      int d = t >> 2, kk = t & 3;
      s = bg[d * 4 + kk];
      for (int i = 0; i < kDin; i++)
        s += xr[d][i] * Wg[((size_t)d * kDin + i) * 4 + kk];
    } else {                            // shared gate: Linear(D_IN, 8)
      int kk = t - 12;
      s = bsg[kk];
      for (int i = 0; i < kDin; i++)
        s += xr[3][i] * Wsg[(size_t)i * 8 + kk];
    }
    lg[t] = s;
  }
  __syncthreads();
  if (t < 4) {
    int base = (t < 3) ? t * 4 : 12;
    int len  = (t < 3) ? 4 : 8;
    float mx = -3.4e38f;
    for (int i = 0; i < len; i++) mx = fmaxf(mx, lg[base + i]);
    float ev[8]; float sum = 0.f;
    for (int i = 0; i < len; i++) { ev[i] = expf(lg[base + i] - mx); sum += ev[i]; }
    float inv = 1.f / sum;
    for (int i = 0; i < len; i++)
      gates[(size_t)row * 20 + base + i] = ev[i] * inv;
  }
}

// ---------------- WMMA GEMM: C = relu(A @ W + bias), f16 in / f32 acc ----
// block = 256 threads (8 waves); block tile 128(M) x 128(N); wave tile 32x64.
// A : [pages][Bsz][K] f16 row-major.
// Wt: [E][N][K] f16 (pre-transposed) -> W tile loader identical to A tile.
// pageMode=1: expert e reads A page (e<6 ? e/2 : 3)   (layer-1 input mapping)
// pageMode=0: expert e reads A page e                 (layer-2: h buffer)
__global__ void __launch_bounds__(256)
gemm_bias_relu_wmma(const _Float16* __restrict__ A,
                    const _Float16* __restrict__ Wt,
                    const float* __restrict__ biasS,
                    const float* __restrict__ biasH,
                    _Float16* __restrict__ C,
                    int Bsz, int K, int N, int pageMode) {
#if USE_ASYNC
  __shared__ alignas(16) uint4 smem4[2048];   // 32 KB: double-buffered (A|W) tiles
#else
  __shared__ alignas(16) uint4 smem4[1024];   // 16 KB: single (A|W) tile
#endif

  const int tid  = threadIdx.x;
  const int lane = tid & 31;
  const int wave = tid >> 5;
  const int waveM = wave & 3;          // 4 waves along M
  const int waveN = wave >> 2;         // 2 waves along N
  const int rl   = lane & 15;
  const int hsel = lane >> 4;          // 0: lanes 0-15, 1: lanes 16-31

  const int e  = blockIdx.z;
  const int n0 = blockIdx.x * 128;
  const int m0 = blockIdx.y * 128;
  const int page = pageMode ? ((e < kSpec) ? (e >> 1) : 3) : e;
  const int ld = K >> 3;               // row stride in uint4 units

  const uint4* __restrict__ Ag = reinterpret_cast<const uint4*>(A) +
      (size_t)page * Bsz * ld + (size_t)m0 * ld;
  const uint4* __restrict__ Wg = reinterpret_cast<const uint4*>(Wt) +
      (size_t)e * N * ld + (size_t)n0 * ld;

  const v8f vzero = {0.f, 0.f, 0.f, 0.f, 0.f, 0.f, 0.f, 0.f};
  v8f acc[2][4];
  #pragma unroll
  for (int mt = 0; mt < 2; mt++)
    #pragma unroll
    for (int nt = 0; nt < 4; nt++) acc[mt][nt] = vzero;

#if USE_ASYNC
  // Fire one (A|W) K-tile into LDS buffer `buf` via ASYNCcnt-tracked DMA.
  auto issue = [&](int k0, int buf) {
    lds_v4i* dA = (lds_v4i*)(smem4 + buf * 1024);
    lds_v4i* dW = dA + 512;
    glb_v4i* sA = (glb_v4i*)(Ag + (k0 >> 3));
    glb_v4i* sW = (glb_v4i*)(Wg + (k0 >> 3));
    #pragma unroll
    for (int i = 0; i < 2; i++) {
      int q4 = tid + i * 256;
      int r = q4 >> 2, c4 = q4 & 3;
      __builtin_amdgcn_global_load_async_to_lds_b128(sA + (size_t)r * ld + c4,
                                                     dA + q4, 0, 0);
      __builtin_amdgcn_global_load_async_to_lds_b128(sW + (size_t)r * ld + c4,
                                                     dW + q4, 0, 0);
    }
  };
  issue(0, 0);
  __builtin_amdgcn_s_wait_asynccnt(0);
  __syncthreads();
#endif

  for (int k0 = 0; k0 < K; k0 += 32) {
#if USE_ASYNC
    const int cur = (k0 >> 5) & 1;
    if (k0 + 32 < K) issue(k0 + 32, cur ^ 1);   // overlap DMA with WMMA
    const uint4* ap = reinterpret_cast<const uint4*>(smem4 + cur * 1024);
    const uint4* bp = ap + 512;
#else
    {  // synchronous fallback: both tiles are plain coalesced b128 copies
      uint4* dA = smem4;
      uint4* dW = smem4 + 512;
      const uint4* sA = Ag + (k0 >> 3);
      const uint4* sW = Wg + (k0 >> 3);
      #pragma unroll
      for (int i = 0; i < 2; i++) {
        int q4 = tid + i * 256;
        int r = q4 >> 2, c4 = q4 & 3;
        dA[q4] = sA[(size_t)r * ld + c4];
        dW[q4] = sW[(size_t)r * ld + c4];
      }
    }
    __syncthreads();
    const uint4* ap = reinterpret_cast<const uint4*>(smem4);
    const uint4* bp = ap + 512;
#endif

    // CDNA5 16-bit A/B fragment layout: lane holds row/col (lane&15);
    // VGPR0-3 = K kb..kb+7, VGPR4-7 = K 16+kb..23+kb, kb = 8*(lane>=16).
    // => exactly two ds_load_b128 per fragment.
    Frag afrag[2], bfrag[4];
    #pragma unroll
    for (int mt = 0; mt < 2; mt++) {
      int r = waveM * 32 + mt * 16 + rl;
      afrag[mt].u[0] = ap[r * 4 + hsel];
      afrag[mt].u[1] = ap[r * 4 + hsel + 2];
    }
    #pragma unroll
    for (int nt = 0; nt < 4; nt++) {
      int c = waveN * 64 + nt * 16 + rl;
      bfrag[nt].u[0] = bp[c * 4 + hsel];
      bfrag[nt].u[1] = bp[c * 4 + hsel + 2];
    }
    #pragma unroll
    for (int mt = 0; mt < 2; mt++)
      #pragma unroll
      for (int nt = 0; nt < 4; nt++)
        acc[mt][nt] = __builtin_amdgcn_wmma_f32_16x16x32_f16(
            false, afrag[mt].h, false, bfrag[nt].h,
            (short)0, acc[mt][nt], false, false);

#if USE_ASYNC
    __builtin_amdgcn_s_wait_asynccnt(0);
#endif
    __syncthreads();
  }

  // ---- epilogue: bias + ReLU, per C/D layout (VGPR j -> M=j / j+8) ----
  const float* bias = (e < kSpec) ? (biasS + (size_t)e * N)
                                  : (biasH + (size_t)(e - kSpec) * N);
  _Float16* Cp = C + (size_t)e * Bsz * N;
  #pragma unroll
  for (int nt = 0; nt < 4; nt++) {
    int col = n0 + waveN * 64 + nt * 16 + rl;
    float bv = bias[col];
    #pragma unroll
    for (int mt = 0; mt < 2; mt++) {
      int rbase = m0 + waveM * 32 + mt * 16 + hsel * 8;
      #pragma unroll
      for (int j = 0; j < 8; j++) {
        float v = acc[mt][nt][j] + bv;
        v = v > 0.f ? v : 0.f;
        Cp[(size_t)(rbase + j) * N + col] = (_Float16)v;
      }
    }
  }
}

// ---------------- gated combine: 4 outputs, each [B, H2] f32 ------------
__global__ void __launch_bounds__(256)
combine_kernel(const _Float16* __restrict__ eo,   // [8][B][H2]
               const float* __restrict__ gates,   // [B][20]
               float* __restrict__ out) {         // [4][B][H2]
  int idx = blockIdx.x * blockDim.x + threadIdx.x;
  if (idx >= kB * kH2) return;
  int b = idx / kH2, o = idx % kH2;
  float ex[8];
  #pragma unroll
  for (int k = 0; k < 8; k++)
    ex[k] = (float)eo[((size_t)k * kB + b) * kH2 + o];
  const float* g = gates + (size_t)b * 20;
  float o0 = g[0] * ex[0] + g[1] * ex[1] + g[2]  * ex[6] + g[3]  * ex[7];
  float o1 = g[4] * ex[2] + g[5] * ex[3] + g[6]  * ex[6] + g[7]  * ex[7];
  float o2 = g[8] * ex[4] + g[9] * ex[5] + g[10] * ex[6] + g[11] * ex[7];
  float o3 = 0.f;
  #pragma unroll
  for (int k = 0; k < 8; k++) o3 += g[12 + k] * ex[k];
  size_t p = (size_t)b * kH2 + o;
  const size_t page = (size_t)kB * kH2;
  out[p]            = o0;
  out[page + p]     = o1;
  out[2 * page + p] = o2;
  out[3 * page + p] = o3;
}

extern "C" void kernel_launch(void* const* d_in, const int* in_sizes, int n_in,
                              void* d_out, int out_size, void* d_ws, size_t ws_size,
                              hipStream_t stream) {
  const float* x0  = (const float*)d_in[0];
  const float* x1  = (const float*)d_in[1];
  const float* x2  = (const float*)d_in[2];
  const float* xs  = (const float*)d_in[3];
  const float* W1s = (const float*)d_in[4];
  const float* b1s = (const float*)d_in[5];
  const float* W2s = (const float*)d_in[6];
  const float* b2s = (const float*)d_in[7];
  const float* W1h = (const float*)d_in[8];
  const float* b1h = (const float*)d_in[9];
  const float* W2h = (const float*)d_in[10];
  const float* b2h = (const float*)d_in[11];
  const float* Wg  = (const float*)d_in[12];
  const float* bg  = (const float*)d_in[13];
  const float* Wsg = (const float*)d_in[14];
  const float* bsg = (const float*)d_in[15];
  (void)in_sizes; (void)n_in; (void)out_size; (void)ws_size;

  // ---- workspace carve-out (~153 MB), 256B aligned ----
  char* ws = (char*)d_ws;
  size_t off = 0;
  auto take = [&](size_t bytes) -> char* {
    char* p = ws + off;
    off += (bytes + 255) & ~(size_t)255;
    return p;
  };
  _Float16* xh  = (_Float16*)take((size_t)4 * kB * kDin * 2);   // x pages f16
  _Float16* w1t = (_Float16*)take((size_t)kE * kH1 * kDin * 2); // W1^T f16 [E][N][K]
  _Float16* w2t = (_Float16*)take((size_t)kE * kH2 * kH1 * 2);  // W2^T f16 [E][N][K]
  _Float16* hb  = (_Float16*)take((size_t)kE * kB * kH1 * 2);   // hidden acts f16
  _Float16* eo  = (_Float16*)take((size_t)kE * kB * kH2 * 2);   // expert outputs f16
  float*    gts = (float*)   take((size_t)kB * 20 * 4);         // gate weights

  auto cvt = [&](const float* s, _Float16* d, size_t n) {
    int n4 = (int)(n >> 2);
    cvt_f32_to_f16<<<(n4 + 255) / 256, 256, 0, stream>>>(s, d, n4);
  };
  cvt(x0, xh + 0 * (size_t)kB * kDin, (size_t)kB * kDin);
  cvt(x1, xh + 1 * (size_t)kB * kDin, (size_t)kB * kDin);
  cvt(x2, xh + 2 * (size_t)kB * kDin, (size_t)kB * kDin);
  cvt(xs, xh + 3 * (size_t)kB * kDin, (size_t)kB * kDin);

  // weights: convert + transpose to [E][N][K] so GEMM streams b128 tiles
  dim3 tb(32, 8);
  cvt_transpose_f32_f16<<<dim3(kH1 / 32, kDin / 32, kSpec), tb, 0, stream>>>(
      W1s, w1t, kDin, kH1);
  cvt_transpose_f32_f16<<<dim3(kH1 / 32, kDin / 32, 2), tb, 0, stream>>>(
      W1h, w1t + (size_t)kSpec * kH1 * kDin, kDin, kH1);
  cvt_transpose_f32_f16<<<dim3(kH2 / 32, kH1 / 32, kSpec), tb, 0, stream>>>(
      W2s, w2t, kH1, kH2);
  cvt_transpose_f32_f16<<<dim3(kH2 / 32, kH1 / 32, 2), tb, 0, stream>>>(
      W2h, w2t + (size_t)kSpec * kH2 * kH1, kH1, kH2);

  gates_kernel<<<kB, 128, 0, stream>>>(x0, x1, x2, xs, Wg, bg, Wsg, bsg, gts);

  // layer 1: h[e] = relu(x_page(e) @ W1[e] + b1[e])   [B,1024]
  dim3 g1(kH1 / 128, kB / 128, kE);
  gemm_bias_relu_wmma<<<g1, 256, 0, stream>>>(xh, w1t, b1s, b1h, hb,
                                              kB, kDin, kH1, /*pageMode=*/1);
  // layer 2: eo[e] = relu(h[e] @ W2[e] + b2[e])       [B,512]
  dim3 g2(kH2 / 128, kB / 128, kE);
  gemm_bias_relu_wmma<<<g2, 256, 0, stream>>>(hb, w2t, b2s, b2h, eo,
                                              kB, kH1, kH2, /*pageMode=*/0);

  combine_kernel<<<(kB * kH2 + 255) / 256, 256, 0, stream>>>(eo, gts, (float*)d_out);
}
